// Net_21689584845427
// MI455X (gfx1250) — matrix operand order repaired and checked by reading, not compiled, not run on hardware
//
#include <hip/hip_runtime.h>
#include <hip/hip_bf16.h>

typedef __attribute__((ext_vector_type(16))) _Float16 v16h;
typedef __attribute__((ext_vector_type(8)))  _Float16 v8h;
typedef __attribute__((ext_vector_type(8)))  float    v8f;

#define IN_DIM 50
#define HIDDEN 128
#define OUT_DIM 121
#define KPAD   128   // padded K stride (halves) for Xh and Wh

// ---------------------------------------------------------------------------
// WMMA GEMM, zero-LDS variant.
// Xh: f16 [Np, KPAD] (rows >= N are zero), Wh: f16 [128 cols][KPAD] (zero pad),
// H : f32 [N, Cout] compact.
// 256 threads = 8 waves; block = 128 rows x 128 cols; wave = 16 rows x 8 tiles.
// Fragments are loaded straight from global memory as contiguous aligned
// vectors (A and B both use contiguous k = 16*(lane/16) + s per lane).
// ---------------------------------------------------------------------------
__global__ __launch_bounds__(256) void gcn_gemm_wmma(
    const _Float16* __restrict__ Xh, const _Float16* __restrict__ Wh,
    float* __restrict__ H, int N, int Kiter, int Cout)
{
  const int lane = threadIdx.x & 31;
  const int wave = threadIdx.x >> 5;                 // 0..7
  const int m    = lane & 15;
  const int g8   = (lane >> 4) << 3;                 // 0 or 8 (halves)
  const int row  = blockIdx.x * 128 + wave * 16 + m; // < Np always

  const _Float16* xrow = Xh + (long)row * KPAD;

  v8f acc[8] = {};

  for (int k0 = 0; k0 < Kiter; k0 += 32) {
    // A-frag: halves 0-15 = K[k0+16g .. +16)
    v8h alo = *(const v8h*)(xrow + k0 + 2 * g8);
    v8h ahi = *(const v8h*)(xrow + k0 + 2 * g8 + 8);
    v16h a = __builtin_shufflevector(alo, ahi,
        0,1,2,3,4,5,6,7,8,9,10,11,12,13,14,15);
    #pragma unroll
    for (int t = 0; t < 8; ++t) {
      // B-frag: halves 0-15 = K[k0+16g .. +16) of column t*16+m
      v16h b = *(const v16h*)(Wh + (long)(t * 16 + m) * KPAD + k0 + 2 * g8);
      acc[t] = __builtin_amdgcn_wmma_f32_16x16x32_f16(
          false, a, false, b, (short)0, acc[t], false, false);
    }
  }

  // Store D: lane<16 -> (M=r, N=lane); lane>=16 -> (M=8+r, N=lane-16)
  const int hi = (lane >> 4) << 3;                   // row offset 0 or 8
  const int rowBase = blockIdx.x * 128 + wave * 16 + hi;
  #pragma unroll
  for (int t = 0; t < 8; ++t) {
    int col = t * 16 + m;
    if (col >= Cout) continue;
    #pragma unroll
    for (int r = 0; r < 8; ++r) {
      int orow = rowBase + r;
      if (orow < N) H[(long)orow * Cout + col] = acc[t][r];
    }
  }
}

// ---------------------------------------------------------------------------
// Prep: x fp32[N,IN_DIM] -> Xh f16[Np,KPAD], zero-padded rows/cols
// ---------------------------------------------------------------------------
__global__ void conv_x(const float* __restrict__ x, _Float16* __restrict__ Xh,
                       int N, long totalPad) {
  long i = (long)blockIdx.x * blockDim.x + threadIdx.x;
  if (i >= totalPad) return;
  int c = (int)(i & (KPAD - 1));
  int r = (int)(i >> 7);
  float v = (r < N && c < IN_DIM) ? x[(long)r * IN_DIM + c] : 0.0f;
  Xh[i] = (_Float16)v;
}

// W fp32[Ci,Co] -> Wh f16 transposed [128][KPAD], zero-padded
__global__ void conv_w(const float* __restrict__ W, _Float16* __restrict__ Wh,
                       int Ci, int Co) {
  int i = blockIdx.x * blockDim.x + threadIdx.x;   // 0 .. 128*128-1
  if (i >= 128 * KPAD) return;
  int k   = i & (KPAD - 1);
  int col = i >> 7;
  float v = (k < Ci && col < Co) ? W[(long)k * Co + col] : 0.0f;
  Wh[i] = (_Float16)v;
}

// ---------------------------------------------------------------------------
// Degree / normalization
// ---------------------------------------------------------------------------
__global__ void init_deg(float* deg, int N) {
  int i = blockIdx.x * blockDim.x + threadIdx.x;
  if (i < N) deg[i] = 1.0f;                        // self loop
}
__global__ void count_deg(const int* __restrict__ dst, float* deg, int E) {
  int e = blockIdx.x * blockDim.x + threadIdx.x;
  if (e < E) unsafeAtomicAdd(&deg[dst[e]], 1.0f);
}
__global__ void rsqrt_deg(float* deg, int N) {
  int i = blockIdx.x * blockDim.x + threadIdx.x;
  if (i < N) deg[i] = rsqrtf(deg[i]);
}
__global__ void zero_buf(float* p, long n) {
  long i = (long)blockIdx.x * blockDim.x + threadIdx.x;
  if (i < n) p[i] = 0.0f;
}

// ---------------------------------------------------------------------------
// Edge scatter: one wave32 per edge (self loops appended).
// Agg[dst, c] += H[src, c] * dinv[src] * dinv[dst]
// ---------------------------------------------------------------------------
__global__ void scatter_edges(const float* __restrict__ H,
                              const int* __restrict__ srcA,
                              const int* __restrict__ dstA,
                              const float* __restrict__ dinv,
                              float* __restrict__ Agg,
                              int E, int N, int C)
{
  long gw  = ((long)blockIdx.x * blockDim.x + threadIdx.x) >> 5;
  int lane = threadIdx.x & 31;
  if (gw >= (long)E + N) return;
  int s, d;
  if (gw < E) { s = srcA[gw]; d = dstA[gw]; }
  else        { s = d = (int)(gw - E); }
  float w = dinv[s] * dinv[d];
  const float* hs = H + (long)s * C;
  float*       ad = Agg + (long)d * C;
  for (int c = lane; c < C; c += 32)
    unsafeAtomicAdd(&ad[c], hs[c] * w);
}

// Hidden layers: Xh[i] = f16(relu(agg[i] + b[c]))   (C == 128 == KPAD)
__global__ void bias_relu_f16(const float* __restrict__ Agg,
                              const float* __restrict__ b,
                              _Float16* __restrict__ Xh, long total)
{
  long i = (long)blockIdx.x * blockDim.x + threadIdx.x;
  if (i >= total) return;
  int c = (int)(i & (KPAD - 1));
  float v = fmaxf(Agg[i] + b[c], 0.0f);
  Xh[i] = (_Float16)v;
}

// Final layer: out[i] = agg[i] + b[i % C]  (pre-relu, fp32)
__global__ void bias_final(const float* __restrict__ Agg,
                           const float* __restrict__ b,
                           float* __restrict__ out, long total, int C)
{
  long i = (long)blockIdx.x * blockDim.x + threadIdx.x;
  if (i >= total) return;
  out[i] = Agg[i] + b[(int)(i % C)];
}

static inline size_t al256(size_t x) { return (x + 255) & ~(size_t)255; }

extern "C" void kernel_launch(void* const* d_in, const int* in_sizes, int n_in,
                              void* d_out, int out_size, void* d_ws, size_t ws_size,
                              hipStream_t stream) {
  const float* x  = (const float*)d_in[0];
  const int*   ei = (const int*)d_in[1];
  const float* Wt[4] = {(const float*)d_in[2], (const float*)d_in[3],
                        (const float*)d_in[4], (const float*)d_in[5]};
  const float* bs[4] = {(const float*)d_in[6], (const float*)d_in[7],
                        (const float*)d_in[8], (const float*)d_in[9]};
  const int N  = in_sizes[0] / IN_DIM;
  const int E  = in_sizes[1] / 2;
  const int Np = ((N + 127) / 128) * 128;
  const int* srcA = ei;
  const int* dstA = ei + E;

  char* ws = (char*)d_ws;
  size_t off = 0;
  _Float16* Xh = (_Float16*)(ws + off); off += al256((size_t)Np * KPAD * sizeof(_Float16));
  _Float16* Wh = (_Float16*)(ws + off); off += al256((size_t)4 * 128 * KPAD * sizeof(_Float16));
  float* bufB  = (float*)(ws + off);    off += al256((size_t)N * HIDDEN * sizeof(float));
  float* bufA  = (float*)(ws + off);    off += al256((size_t)N * HIDDEN * sizeof(float));
  float* deg   = (float*)(ws + off);    off += al256((size_t)N * sizeof(float));
  (void)ws_size; (void)n_in; (void)out_size;

  const dim3 blk(256);

  // --- one-time prep: f16 conversions + normalization coefficients ---
  const long xhTot = (long)Np * KPAD;
  conv_x<<<(int)((xhTot + 255) / 256), blk, 0, stream>>>(x, Xh, N, xhTot);
  const int dims_in[4]  = {IN_DIM, HIDDEN, HIDDEN, HIDDEN};
  const int dims_out[4] = {HIDDEN, HIDDEN, HIDDEN, OUT_DIM};
  for (int l = 0; l < 4; ++l)
    conv_w<<<(128 * KPAD) / 256, blk, 0, stream>>>(Wt[l], Wh + (size_t)l * 128 * KPAD,
                                                   dims_in[l], dims_out[l]);
  init_deg <<<(N + 255) / 256, blk, 0, stream>>>(deg, N);
  count_deg<<<(E + 255) / 256, blk, 0, stream>>>(dstA, deg, E);
  rsqrt_deg<<<(N + 255) / 256, blk, 0, stream>>>(deg, N);

  const long totalWaves = (long)E + N;
  const int scatterBlocks = (int)((totalWaves * 32 + 255) / 256);
  const int gemmBlocks    = Np / 128;
  const int kIter[4] = {64, 128, 128, 128};   // layer 0: K=50 padded to 64

  for (int l = 0; l < 4; ++l) {
    const int Co = dims_out[l];
    const long tot = (long)N * Co;

    // h = X @ W  (WMMA, fragments straight from global f16)
    gcn_gemm_wmma<<<gemmBlocks, blk, 0, stream>>>(
        Xh, Wh + (size_t)l * 128 * KPAD, bufB, N, kIter[l], Co);

    // agg = segment_sum(h[src] * norm)
    zero_buf<<<(int)((tot + 255) / 256), blk, 0, stream>>>(bufA, tot);
    scatter_edges<<<scatterBlocks, blk, 0, stream>>>(bufB, srcA, dstA, deg,
                                                     bufA, E, N, Co);

    if (l < 3) {
      bias_relu_f16<<<(int)((tot + 255) / 256), blk, 0, stream>>>(bufA, bs[l], Xh, tot);
    } else {
      bias_final<<<(int)((tot + 255) / 256), blk, 0, stream>>>(bufA, bs[l],
                                                               (float*)d_out, tot, Co);
    }
  }
}
